// MambaModel_3624952398807
// MI455X (gfx1250) — matrix operand order repaired and checked by reading, not compiled, not run on hardware
//
#include <hip/hip_runtime.h>
#include <hip/hip_bf16.h>
#include <stdint.h>

#define V_SIZE   32000
#define D_MODEL  1024
#define N_LAYER  2
#define D_STATE  16
#define D_CONV   4
#define D_INNER  2048
#define DT_RANK  64
#define BATCH    2
#define SEQ      2048
#define NTOK     (BATCH * SEQ)
#define XDBL_C   (DT_RANK + 2 * D_STATE)   // 96

typedef _Float16     v8h  __attribute__((ext_vector_type(8)));
typedef _Float16     v16h __attribute__((ext_vector_type(16)));
typedef float        v8f  __attribute__((ext_vector_type(8)));
typedef unsigned int v4u  __attribute__((ext_vector_type(4)));
typedef int          v8i  __attribute__((ext_vector_type(8)));
typedef int          v4i  __attribute__((ext_vector_type(4)));

union F16Frag { v16h v; v8h h[2]; };

__device__ __forceinline__ float silu_f(float x) { return x / (1.0f + __expf(-x)); }

// ---------------------------------------------------------------------------
// Generic GEMM: C[M,N] = A[M,K] * W[N,K]^T   (A,W f16 row-major, C f32)
// Block: 256 threads (8 wave32), tile 128x128, K step 32.
// W tile staged into LDS by the Tensor Data Mover (TDM); A tile by vector loads.
// ---------------------------------------------------------------------------
#define TM 128
#define TN 128
#define TK 32

__global__ __launch_bounds__(256) void gemm_wmma_f16(
    const _Float16* __restrict__ A,
    const _Float16* __restrict__ W,
    float* __restrict__ C,
    int M, int N, int K)
{
  __shared__ __align__(16) _Float16 As[TM][TK];
  __shared__ __align__(16) _Float16 Ws[TN][TK];

  const int tid  = threadIdx.x;
  const int lane = tid & 31;
  const int wid  = tid >> 5;      // 0..7
  const int wm   = wid >> 2;      // 0..1 -> which 64-row half
  const int wn   = wid & 3;       // 0..3 -> which 32-col quarter
  const int m0   = blockIdx.y * TM;
  const int n0   = blockIdx.x * TN;

  v8f zero = {};
  v8f acc[4][2];
  #pragma unroll
  for (int i = 0; i < 4; ++i)
    #pragma unroll
    for (int j = 0; j < 2; ++j) acc[i][j] = zero;

  for (int k0 = 0; k0 < K; k0 += TK) {
    __syncthreads();
    // ---- stage A tile: 128 rows x 32 halfs (8 KB), 16B chunks -------------
    #pragma unroll
    for (int it = 0; it < 2; ++it) {
      int idx = tid + it * 256;          // 0..511
      int row = idx >> 2;
      int ch  = idx & 3;
      const v8h* gp = (const v8h*)(A + (size_t)(m0 + row) * K + k0 + ch * 8);
      *(v8h*)&As[row][ch * 8] = *gp;
    }
    // ---- stage W tile via Tensor Data Mover (wave 0 issues the DMA) -------
#if defined(__gfx1250__) && __has_builtin(__builtin_amdgcn_tensor_load_to_lds)
    if (wid == 0) {
      unsigned long long ga = (unsigned long long)(uintptr_t)(W + (size_t)n0 * K + k0);
      unsigned lds_off = (unsigned)(uintptr_t)&Ws[0][0];
      v4u g0;
      g0[0] = 1u;                                               // count=1, user mode
      g0[1] = lds_off;                                          // lds_addr
      g0[2] = (unsigned)(ga & 0xFFFFFFFFull);                   // global_addr lo
      g0[3] = (unsigned)((ga >> 32) & 0x01FFFFFFull) | (2u << 30); // hi | type=2
      v8i g1;
      g1[0] = (int)(1u << 16);                                  // data_size = 2B
      g1[1] = (int)(((unsigned)K & 0xFFFFu) << 16);             // tensor_dim0 lo16
      g1[2] = (int)(((unsigned)K >> 16) | (((unsigned)N & 0xFFFFu) << 16));
      g1[3] = (int)(((unsigned)N >> 16) | (32u << 16));         // tile_dim0 = 32
      g1[4] = 128;                                              // tile_dim1=128, tile_dim2=0
      g1[5] = (int)(unsigned)K;                                 // tensor_dim0_stride lo
      g1[6] = 0;
      g1[7] = 0;
      v4i gz = {};
      v8i gz8 = {};
      __builtin_amdgcn_tensor_load_to_lds(g0, g1, gz, gz, gz8, 0);
      __builtin_amdgcn_s_wait_tensorcnt(0);
    }
#else
    #pragma unroll
    for (int it = 0; it < 2; ++it) {
      int idx = tid + it * 256;
      int row = idx >> 2;
      int ch  = idx & 3;
      int gn  = n0 + row;
      v8h val = {};
      if (gn < N) val = *(const v8h*)(W + (size_t)gn * K + k0 + ch * 8);
      *(v8h*)&Ws[row][ch * 8] = val;
    }
#endif
    __syncthreads();

    // ---- WMMA compute: each wave 4x2 tiles of 16x16 -----------------------
    #pragma unroll
    for (int mi = 0; mi < 4; ++mi) {
      F16Frag a;
      int ar  = wm * 64 + mi * 16 + (lane & 15);
      int akb = (lane & 16) ? 8 : 0;            // ISA 16-bit A layout
      a.h[0] = *(const v8h*)&As[ar][akb];
      a.h[1] = *(const v8h*)&As[ar][akb + 16];
      #pragma unroll
      for (int ni = 0; ni < 2; ++ni) {
        F16Frag b;
        int wr  = wn * 32 + ni * 16 + (lane & 15);
        int wkb = (lane & 16) ? 8 : 0;          // ISA 16-bit B layout
        b.h[0] = *(const v8h*)&Ws[wr][wkb];
        b.h[1] = *(const v8h*)&Ws[wr][wkb + 16];
        acc[mi][ni] = __builtin_amdgcn_wmma_f32_16x16x32_f16(
            false, a.v, false, b.v, (short)0, acc[mi][ni], false, false);
      }
    }
  }

  // ---- store: D layout => VGPR r holds row (r + 8*(lane>=16)), col lane%16
  #pragma unroll
  for (int mi = 0; mi < 4; ++mi) {
    #pragma unroll
    for (int ni = 0; ni < 2; ++ni) {
      int rbase = m0 + wm * 64 + mi * 16 + ((lane & 16) ? 8 : 0);
      int col   = n0 + wn * 32 + ni * 16 + (lane & 15);
      if (col < N) {
        #pragma unroll
        for (int r = 0; r < 8; ++r)
          C[(size_t)(rbase + r) * N + col] = acc[mi][ni][r];
      }
    }
  }
}

// ---------------------------------------------------------------------------
// Elementwise / model kernels
// ---------------------------------------------------------------------------
__global__ void embed_kernel(const int* __restrict__ tokens,
                             const float* __restrict__ emb,
                             float* __restrict__ x)
{
  int t = blockIdx.x;
  int tok = tokens[t];
  const float* src = emb + (size_t)tok * D_MODEL;
  float* dst = x + (size_t)t * D_MODEL;
  for (int i = threadIdx.x; i < D_MODEL; i += blockDim.x) dst[i] = src[i];
}

__global__ void cvt_f16_kernel(const float* __restrict__ src, _Float16* __restrict__ dst,
                               long long n, int cols, int src_stride)
{
  long long i = (long long)blockIdx.x * blockDim.x + threadIdx.x;
  if (i >= n) return;
  long long r = i / cols;
  int c = (int)(i % cols);
  dst[i] = (_Float16)src[r * (long long)src_stride + c];
}

__global__ void conv_silu_kernel(const float* __restrict__ xz,
                                 const float* __restrict__ cw,
                                 const float* __restrict__ cb,
                                 float* __restrict__ u)
{
  long long i = (long long)blockIdx.x * blockDim.x + threadIdx.x;
  if (i >= (long long)NTOK * D_INNER) return;
  int c = (int)(i % D_INNER);
  long long tok = i / D_INNER;
  int l = (int)(tok % SEQ);
  float acc = cb[c];
  #pragma unroll
  for (int k = 0; k < D_CONV; ++k) {
    int ll = l - (D_CONV - 1) + k;
    if (ll >= 0)
      acc += xz[(tok - l + ll) * (2 * D_INNER) + c] * cw[c * D_CONV + k];
  }
  u[i] = silu_f(acc);
}

__global__ void softplus_bias_kernel(float* __restrict__ delta,
                                     const float* __restrict__ bias)
{
  long long i = (long long)blockIdx.x * blockDim.x + threadIdx.x;
  if (i >= (long long)NTOK * D_INNER) return;
  int c = (int)(i % D_INNER);
  float d = delta[i] + bias[c];
  delta[i] = (d > 20.f) ? d : log1pf(__expf(d));
}

// Sequential selective scan: one channel per thread; 16 states in registers.
__global__ __launch_bounds__(256) void scan_kernel(
    const float* __restrict__ u, const float* __restrict__ delta,
    const float* __restrict__ xdbl, const float* __restrict__ A_log,
    float* __restrict__ yssm)
{
  __shared__ float sBC[2 * D_STATE];
  int bc = blockIdx.x * blockDim.x + threadIdx.x;  // 0 .. BATCH*D_INNER-1
  int b = bc >> 11;                                // D_INNER == 2048
  int c = bc & (D_INNER - 1);
  float Ac[D_STATE], h[D_STATE];
  #pragma unroll
  for (int s = 0; s < D_STATE; ++s) {
    Ac[s] = -__expf(A_log[(size_t)c * D_STATE + s]);
    h[s]  = 0.f;
  }
  for (int t = 0; t < SEQ; ++t) {
    long long tok = (long long)b * SEQ + t;
    if (threadIdx.x < 2 * D_STATE)
      sBC[threadIdx.x] = xdbl[tok * XDBL_C + DT_RANK + threadIdx.x];
    __syncthreads();
    float d  = delta[tok * D_INNER + c];
    float ut = u[tok * D_INNER + c];
    float du = d * ut;
    float y = 0.f;
    #pragma unroll
    for (int s = 0; s < D_STATE; ++s) {
      h[s] = h[s] * __expf(d * Ac[s]) + du * sBC[s];
      y += h[s] * sBC[D_STATE + s];
    }
    yssm[tok * D_INNER + c] = y;
    __syncthreads();
  }
}

__global__ void gate_kernel(float* __restrict__ y, const float* __restrict__ u,
                            const float* __restrict__ xz,
                            const float* __restrict__ Dskip)
{
  long long i = (long long)blockIdx.x * blockDim.x + threadIdx.x;
  if (i >= (long long)NTOK * D_INNER) return;
  int c = (int)(i % D_INNER);
  long long tok = i / D_INNER;
  float z = xz[tok * (2 * D_INNER) + D_INNER + c];
  y[i] = (y[i] + u[i] * Dskip[c]) * silu_f(z);
}

__global__ __launch_bounds__(256) void layernorm_f16_kernel(
    const float* __restrict__ x, const float* __restrict__ gamma,
    const float* __restrict__ beta, _Float16* __restrict__ out)
{
  __shared__ float red[256];
  const int t = blockIdx.x;
  const float* row = x + (size_t)t * D_MODEL;
  float s = 0.f;
  for (int i = threadIdx.x; i < D_MODEL; i += 256) s += row[i];
  red[threadIdx.x] = s; __syncthreads();
  for (int o = 128; o > 0; o >>= 1) {
    if (threadIdx.x < o) red[threadIdx.x] += red[threadIdx.x + o];
    __syncthreads();
  }
  float mu = red[0] / (float)D_MODEL;
  __syncthreads();
  float v = 0.f;
  for (int i = threadIdx.x; i < D_MODEL; i += 256) { float d = row[i] - mu; v += d * d; }
  red[threadIdx.x] = v; __syncthreads();
  for (int o = 128; o > 0; o >>= 1) {
    if (threadIdx.x < o) red[threadIdx.x] += red[threadIdx.x + o];
    __syncthreads();
  }
  float rstd = rsqrtf(red[0] / (float)D_MODEL + 1e-5f);
  _Float16* orow = out + (size_t)t * D_MODEL;
  for (int i = threadIdx.x; i < D_MODEL; i += 256)
    orow[i] = (_Float16)((row[i] - mu) * rstd * gamma[i] + beta[i]);
}

// ---------------------------------------------------------------------------
// Host orchestration
// ---------------------------------------------------------------------------
static inline size_t alignup256(size_t x) { return (x + 255) & ~(size_t)255; }

extern "C" void kernel_launch(void* const* d_in, const int* in_sizes, int n_in,
                              void* d_out, int out_size, void* d_ws, size_t ws_size,
                              hipStream_t stream)
{
  (void)in_sizes; (void)n_in; (void)out_size; (void)ws_size;

  const int*   tokens     = (const int*)  d_in[0];
  const float* embedding  = (const float*)d_in[1];
  const float* in_proj_w  = (const float*)d_in[2];
  const float* conv_w     = (const float*)d_in[3];
  const float* conv_b     = (const float*)d_in[4];
  const float* x_proj_w   = (const float*)d_in[5];
  const float* dt_proj_w  = (const float*)d_in[6];
  const float* dt_proj_b  = (const float*)d_in[7];
  const float* A_log      = (const float*)d_in[8];
  const float* D_skip     = (const float*)d_in[9];
  const float* out_proj_w = (const float*)d_in[10];
  const float* norm_gamma = (const float*)d_in[11];
  const float* norm_beta  = (const float*)d_in[12];
  const float* lm_head_w  = (const float*)d_in[13];
  float* logits = (float*)d_out;

  char* p = (char*)d_ws;
  auto carve = [&](size_t bytes) { char* r = p; p += alignup256(bytes); return r; };
  float*    x      = (float*)   carve((size_t)NTOK * D_MODEL * 4);
  float*    xz     = (float*)   carve((size_t)NTOK * 2 * D_INNER * 4);
  float*    u      = (float*)   carve((size_t)NTOK * D_INNER * 4);
  float*    xdbl   = (float*)   carve((size_t)NTOK * XDBL_C * 4);
  float*    delta  = (float*)   carve((size_t)NTOK * D_INNER * 4);
  float*    yssm   = (float*)   carve((size_t)NTOK * D_INNER * 4);
  _Float16* actA   = (_Float16*)carve((size_t)NTOK * D_INNER * 2);
  _Float16* w_in   = (_Float16*)carve((size_t)N_LAYER * 2 * D_INNER * D_MODEL * 2);
  _Float16* w_xp   = (_Float16*)carve((size_t)N_LAYER * XDBL_C * D_INNER * 2);
  _Float16* w_dt   = (_Float16*)carve((size_t)N_LAYER * D_INNER * DT_RANK * 2);
  _Float16* w_out  = (_Float16*)carve((size_t)N_LAYER * D_MODEL * D_INNER * 2);
  _Float16* w_head = (_Float16*)carve((size_t)V_SIZE * D_MODEL * 2);

  auto cvt = [&](const float* src, _Float16* dst, long long rows, int cols, int stride) {
    long long n = rows * (long long)cols;
    int blocks = (int)((n + 255) / 256);
    cvt_f16_kernel<<<blocks, 256, 0, stream>>>(src, dst, n, cols, stride);
  };

  // f16 copies of all weights (deterministic, every call)
  cvt(in_proj_w,  w_in,   (long long)N_LAYER * 2 * D_INNER, D_MODEL, D_MODEL);
  cvt(x_proj_w,   w_xp,   (long long)N_LAYER * XDBL_C,      D_INNER, D_INNER);
  cvt(dt_proj_w,  w_dt,   (long long)N_LAYER * D_INNER,     DT_RANK, DT_RANK);
  cvt(out_proj_w, w_out,  (long long)N_LAYER * D_MODEL,     D_INNER, D_INNER);
  cvt(lm_head_w,  w_head, (long long)V_SIZE,                D_MODEL, D_MODEL);

  embed_kernel<<<NTOK, 256, 0, stream>>>(tokens, embedding, x);

  const long long nEW = (long long)NTOK * D_INNER;
  const int ewBlocks = (int)((nEW + 255) / 256);
  dim3 blk(256);

  for (int layer = 0; layer < N_LAYER; ++layer) {
    const _Float16* win  = w_in  + (size_t)layer * 2 * D_INNER * D_MODEL;
    const _Float16* wxp  = w_xp  + (size_t)layer * XDBL_C * D_INNER;
    const _Float16* wdt  = w_dt  + (size_t)layer * D_INNER * DT_RANK;
    const _Float16* wout = w_out + (size_t)layer * D_MODEL * D_INNER;

    // xz = x @ in_w^T
    cvt(x, actA, NTOK, D_MODEL, D_MODEL);
    gemm_wmma_f16<<<dim3((2 * D_INNER) / TN, NTOK / TM), blk, 0, stream>>>(
        actA, win, xz, NTOK, 2 * D_INNER, D_MODEL);

    // u = silu(causal_dwconv(xz[:, :d_inner]))
    conv_silu_kernel<<<ewBlocks, blk, 0, stream>>>(
        xz, conv_w + (size_t)layer * D_INNER * D_CONV,
        conv_b + (size_t)layer * D_INNER, u);

    // x_dbl = u @ xp_w^T   (N=96 edge handled by tensor-dim OOB zeros + store guard)
    cvt(u, actA, NTOK, D_INNER, D_INNER);
    gemm_wmma_f16<<<dim3(1, NTOK / TM), blk, 0, stream>>>(
        actA, wxp, xdbl, NTOK, XDBL_C, D_INNER);

    // delta = softplus(dt @ dt_w^T + dt_b); dt = x_dbl[:, :64] (strided convert)
    cvt(xdbl, actA, NTOK, DT_RANK, XDBL_C);
    gemm_wmma_f16<<<dim3(D_INNER / TN, NTOK / TM), blk, 0, stream>>>(
        actA, wdt, delta, NTOK, D_INNER, DT_RANK);
    softplus_bias_kernel<<<ewBlocks, blk, 0, stream>>>(
        delta, dt_proj_b + (size_t)layer * D_INNER);

    // selective scan
    scan_kernel<<<(BATCH * D_INNER) / 256, blk, 0, stream>>>(
        u, delta, xdbl, A_log + (size_t)layer * D_INNER * D_STATE, yssm);

    // y = (ys + u*D) * silu(z)
    gate_kernel<<<ewBlocks, blk, 0, stream>>>(
        yssm, u, xz, D_skip + (size_t)layer * D_INNER);

    // x = y @ out_w^T
    cvt(yssm, actA, NTOK, D_INNER, D_INNER);
    gemm_wmma_f16<<<dim3(D_MODEL / TN, NTOK / TM), blk, 0, stream>>>(
        actA, wout, x, NTOK, D_MODEL, D_INNER);
  }

  // LayerNorm -> f16, then logits = xn @ lm_head^T
  layernorm_f16_kernel<<<NTOK, blk, 0, stream>>>(x, norm_gamma, norm_beta, actA);
  gemm_wmma_f16<<<dim3(V_SIZE / TN, NTOK / TM), blk, 0, stream>>>(
      actA, w_head, logits, NTOK, V_SIZE, D_MODEL);
}
